// MACDUnidirectional_17308718203419
// MI455X (gfx1250) — matrix operand order repaired and checked by reading, not compiled, not run on hardware
//
#include <hip/hip_runtime.h>

// ---------------------------------------------------------------------------
// DAMSM loss (AttnGAN) for MI455X / gfx1250.
// B=64, S=30, E=256, HW*HW=289.
// GEMMs: v_wmma_f32_16x16x32_bf16.  LDS staging: TDM tensor_load_to_lds from
// pre-packed bf16 copies of img (made once by a pack kernel into d_ws).
// LDS = 151 KB -> two workgroups co-resident per 320 KB WGP.
// ---------------------------------------------------------------------------

#define NB   64    // batch (images == sentences)
#define NS   30    // words per sentence
#define NE   256   // embedding dim
#define HW2  289   // 17*17 pixels
#define SP   304   // s padded to 19*16 for the scores buffer
#define SP2  320   // s padded to 20*16 for phase-1 tiles (TDM zero-fills OOB)
#define KP   320   // s padded to 10*32 for GEMM2 K-steps
#define QP   32    // q padded to 2*16

#define GAMMA1 4.0f
#define GAMMA2 5.0f
#define GAMMA3 10.0f

typedef __attribute__((ext_vector_type(16))) __bf16 v16bf;
typedef __attribute__((ext_vector_type(8)))  __bf16 v8bf;
typedef __attribute__((ext_vector_type(8)))  float  v8f;
typedef __attribute__((ext_vector_type(4)))  unsigned int v4u;
typedef __attribute__((ext_vector_type(8)))  unsigned int v8u;

__device__ __forceinline__ v8f wmma_bf16(v16bf a, v16bf b, v8f c) {
  // (neg_a, A, neg_b, B, c_mod, C, reuse_a, reuse_b)
  return __builtin_amdgcn_wmma_f32_16x16x32_bf16(false, a, false, b,
                                                 (short)0, c, false, false);
}

__device__ __forceinline__ v8f zero_v8f() {
  v8f z;
#pragma unroll
  for (int k = 0; k < 8; ++k) z[k] = 0.f;
  return z;
}

// A-frag k order used here (contiguous, as B): lane<16 -> K = {khA..khA+15}, khA=0;
// lane>=16 -> khA=16.  Two 16B LDS chunks.
__device__ __forceinline__ v16bf load_a_frag(const __bf16* base, int khA) {
  v8bf lo = *(const v8bf*)(base + khA);
  v8bf hi = *(const v8bf*)(base + khA + 8);
  v16bf a;
#pragma unroll
  for (int j = 0; j < 8; ++j) { a[j] = lo[j]; a[j + 8] = hi[j]; }
  return a;
}

// ---------------------------------------------------------------------------
// TDM: 2D tile load from global to LDS (bf16 elements, data_size=2).
//   tensor: dim0 (contiguous, elems), dim1 (rows), row stride stride0 (elems)
//   tile:   tile0 x tile1, starting at gptr; packed contiguously at lds_off.
//   Tile rows/cols beyond the tensor dims are zero-filled by the TDM.
// D# layout per CDNA5 ISA 8.3/8.4 (count=1, type=2 "image").
// ---------------------------------------------------------------------------
__device__ __forceinline__ void tdm_load_2d(unsigned lds_off, const void* gptr,
                                            unsigned dim0, unsigned dim1,
                                            unsigned stride0,
                                            unsigned tile0, unsigned tile1) {
  unsigned long long ga = (unsigned long long)(uintptr_t)gptr;
  v4u g0;
  g0[0] = 1u;                                        // count=1, user mode
  g0[1] = lds_off;                                   // lds_addr (bytes)
  g0[2] = (unsigned)ga;                              // global_addr[31:0]
  g0[3] = ((unsigned)(ga >> 32) & 0x01FFFFFFu)       // global_addr[56:32]
          | (2u << 30);                              // type = 2
  v8u g1;
  g1[0] = 1u << 16;                                  // data_size=1 (2 bytes)
  g1[1] = (dim0 & 0xFFFFu) << 16;                    // tensor_dim0[15:0]
  g1[2] = (dim0 >> 16) | ((dim1 & 0xFFFFu) << 16);   // dim0 hi | dim1 lo
  g1[3] = (dim1 >> 16) | ((tile0 & 0xFFFFu) << 16);  // dim1 hi | tile_dim0
  g1[4] = tile1 & 0xFFFFu;                           // tile_dim1 (tile_dim2=0)
  g1[5] = stride0;                                   // tensor_dim0_stride lo
  g1[6] = 0u;                                        // stride0 hi | stride1 lo
  g1[7] = 0u;                                        // stride1 hi
  asm volatile("tensor_load_to_lds %0, %1" :: "s"(g0), "s"(g1) : "memory");
}

// ---------------------------------------------------------------------------
// Pack kernel: img fp32 [b][e][289] -> two bf16 copies in workspace:
//   ctxT[b][s=304][e=256]  (B-operand for GEMM1, zero-padded s)
//   ctxR[b][e=256][s=320]  (A-operand for GEMM2, zero-padded s)
// ---------------------------------------------------------------------------
extern "C" __global__ __launch_bounds__(256)
void damsm_pack_kernel(const float* __restrict__ img,
                       __bf16* __restrict__ ctxT,
                       __bf16* __restrict__ ctxR) {
  const int b = blockIdx.x;
  const float* src = img + (size_t)b * NE * HW2;
  __bf16* dT = ctxT + (size_t)b * SP * NE;
  __bf16* dR = ctxR + (size_t)b * NE * KP;
  for (int idx = threadIdx.x; idx < NE * KP; idx += 256) {
    int e = idx / KP, s = idx % KP;
    dR[idx] = (s < HW2) ? (__bf16)src[e * HW2 + s] : (__bf16)0.f;
  }
  for (int idx = threadIdx.x; idx < SP * NE; idx += 256) {
    int s = idx / NE, e = idx % NE;
    dT[idx] = (s < HW2) ? (__bf16)src[e * HW2 + s] : (__bf16)0.f;
  }
}

// Dynamic LDS layout (bytes):
//   Wf      float [32][256]   32768   @      0
//   scores  float [304][32]   38912   @  32768
//   ctxb    bf16  [20480]     40960   @  71680  (phase1 [320][64] / ph3 [64][320])
//   attnT   bf16  [32][320]   20480   @ 112640
//   Wbf     bf16  [32][256]   16384   @ 133120
//   red     float [8][32]      1024   @ 149504
//   qmax,qsum,w12,w2sq,smask: 5*32*4 = 640
#define SMEM_BYTES (32768 + 38912 + 40960 + 20480 + 16384 + 1024 + 640)

extern "C" __global__ __launch_bounds__(256)
void damsm_word_kernel(const __bf16* __restrict__ ctxT,   // [64][304][256]
                       const __bf16* __restrict__ ctxR,   // [64][256][320]
                       const float*  __restrict__ words,  // [64][30][256]
                       const float*  __restrict__ smask_g,// [64][30]
                       float* __restrict__ simT)          // [b_img][i_sent]
{
  extern __shared__ char smem_raw[];
  float*  Wf     = (float*)smem_raw;                 // 32*256
  float*  scores = Wf + QP * NE;                     // 304*32
  __bf16* ctxb   = (__bf16*)(scores + SP * QP);      // 20480 elems
  __bf16* attnT  = ctxb + 64 * KP;                   // 32*320 elems
  __bf16* Wbf    = attnT + QP * KP;                  // 32*256 elems
  float*  red    = (float*)(Wbf + QP * NE);          // 8*32
  float*  qmax   = red + 8 * QP;
  float*  qsum   = qmax + QP;
  float*  w12    = qsum + QP;
  float*  w2sq   = w12 + QP;
  float*  smask  = w2sq + QP;

  const int tid    = threadIdx.x;
  const int lane   = tid & 31;
  const int wave   = tid >> 5;
  const int i_sent = blockIdx.x >> 6;
  const int b_img  = blockIdx.x & 63;
  const __bf16* ctxTb = ctxT + (size_t)b_img * SP * NE;
  const __bf16* ctxRb = ctxR + (size_t)b_img * NE * KP;
  const unsigned ctxb_lds = (unsigned)(uintptr_t)ctxb;

  // ---------------- Phase 0: stage words[i] (fp32 + bf16) + mask ------------
  for (int idx = tid; idx < QP * NE; idx += 256) {
    int q = idx >> 8, e = idx & 255;
    float v = (q < NS) ? words[((size_t)i_sent * NS + q) * NE + e] : 0.f;
    Wf[idx]  = v;
    Wbf[idx] = (__bf16)v;
  }
  if (tid < QP) {
    smask[tid] = (tid < NS) ? smask_g[i_sent * NS + tid] : 0.f;
    w12[tid] = 0.f;
    w2sq[tid] = 0.f;
  }
  __syncthreads();

  // ---------------- Phase 1: scores[q][s] = W(32x256) x ctx(256x289) --------
  // 40 tiles = 2 q-tiles x 20 s-tiles; wave w owns fixed q-tile (w&1) and
  // s-tiles (w>>1)+4m -> one A-frag feeds 5 WMMAs per k-step, no divergence.
  v8f acc[5];
#pragma unroll
  for (int m = 0; m < 5; ++m) acc[m] = zero_v8f();

  const int khA = (lane < 16) ? 0 : 16;  // A-frag K base (contiguous k order)
  const int koB = (lane < 16) ? 0 : 16;  // B-frag K base
  const int qt  = wave & 1;
  const int st0 = wave >> 1;
  const int qrow = qt * 16 + (lane & 15);

  for (int ch = 0; ch < 4; ++ch) {       // chunks of 64 over e (=K)
    // TDM: tile [s=0..319][e = ch*64 .. +63] from ctxT -> LDS [320][64]
    // (rows 304..319 are beyond tensor_dim1=304 -> zero-filled)
    if (tid == 0)
      tdm_load_2d(ctxb_lds, ctxTb + ch * 64, NE, SP, NE, 64, SP2);
    __builtin_amdgcn_s_wait_tensorcnt(0);
    __syncthreads();

#pragma unroll
    for (int kk = 0; kk < 2; ++kk) {
      v16bf a = load_a_frag(&Wbf[qrow * NE + ch * 64 + kk * 32], khA);
#pragma unroll
      for (int m = 0; m < 5; ++m) {
        int srow = (st0 + 4 * m) * 16 + (lane & 15);
        v16bf b = *(const v16bf*)&ctxb[srow * 64 + kk * 32 + koB];
        acc[m] = wmma_bf16(a, b, acc[m]);
      }
    }
    __syncthreads();
  }

  // write accumulators to scores[s][q] (skip the padded s-tile 19)
  {
    int s16 = lane & 15;
    int qb  = qt * 16 + ((lane < 16) ? 0 : 8);
#pragma unroll
    for (int m = 0; m < 5; ++m) {
      int st = st0 + 4 * m;
      if (st < 19) {
        int s = st * 16 + s16;
#pragma unroll
        for (int r = 0; r < 8; ++r) scores[s * QP + qb + r] = acc[m][r];
      }
    }
  }
  __syncthreads();

  // ---------------- Phase 2a: masked softmax over q, scale by GAMMA1 --------
  for (int s = tid; s < HW2; s += 256) {
    float mx = -1e30f;
    for (int q = 0; q < NS; ++q)
      if (smask[q] > 0.5f) mx = fmaxf(mx, scores[s * QP + q]);
    float sum = 0.f;
    for (int q = 0; q < NS; ++q)
      if (smask[q] > 0.5f) sum += __expf(scores[s * QP + q] - mx);
    float inv = GAMMA1 / sum;
    for (int q = 0; q < QP; ++q) {
      float v = (q < NS && smask[q] > 0.5f)
                    ? __expf(scores[s * QP + q] - mx) * inv : 0.f;
      scores[s * QP + q] = v;
    }
  }
  __syncthreads();

  // ---------------- Phase 2b: softmax over s (289) per q; write attnT bf16 --
  {
    int q = tid & 31, p = tid >> 5;  // 8 partials per q
    float pm = -1e30f;
    for (int s = p; s < HW2; s += 8) pm = fmaxf(pm, scores[s * QP + q]);
    red[p * QP + q] = pm;
    __syncthreads();
    if (p == 0) {
      float m = red[q];
      for (int k = 1; k < 8; ++k) m = fmaxf(m, red[k * QP + q]);
      qmax[q] = m;
    }
    __syncthreads();
    float ps = 0.f;
    for (int s = p; s < HW2; s += 8) ps += __expf(scores[s * QP + q] - qmax[q]);
    red[p * QP + q] = ps;
    __syncthreads();
    if (p == 0) {
      float sm = 0.f;
      for (int k = 0; k < 8; ++k) sm += red[k * QP + q];
      qsum[q] = sm;
    }
    __syncthreads();
    float inv = 1.f / qsum[q];
    for (int s = p; s < KP; s += 8) {
      float v = (s < HW2) ? __expf(scores[s * QP + q] - qmax[q]) * inv : 0.f;
      attnT[q * KP + s] = (__bf16)v;
    }
  }
  __syncthreads();

  // ---------------- Phase 3: wei_ctx[e][q] = ctx(256x289) x attn(289x32) ----
  // per e-chunk: 8 tiles (4 e-tiles x 2 q-tiles), one per wave; K = 320
  {
    int et = wave >> 1, qt3 = wave & 1;
    int erow = et * 16 + (lane & 15);
    int qcol = qt3 * 16 + (lane & 15);
    for (int ch = 0; ch < 4; ++ch) {
      // TDM: tile [e = ch*64 .. +63][s=0..319] from ctxR -> LDS [64][320]
      if (tid == 0)
        tdm_load_2d(ctxb_lds, ctxRb + (size_t)ch * 64 * KP, KP, NE, KP, KP, 64);
      __builtin_amdgcn_s_wait_tensorcnt(0);
      __syncthreads();

      v8f acc2 = zero_v8f();
#pragma unroll
      for (int kk = 0; kk < 10; ++kk) {
        v16bf a = load_a_frag(&ctxb[erow * KP + kk * 32], khA);
        v16bf b = *(const v16bf*)&attnT[qcol * KP + kk * 32 + koB];
        acc2 = wmma_bf16(a, b, acc2);
      }
      // fold wei_ctx chunk into w12[q], w2sq[q] directly from D layout
      float s1 = 0.f, s2 = 0.f;
#pragma unroll
      for (int r = 0; r < 8; ++r) {
        int eg = ch * 64 + et * 16 + ((lane < 16) ? 0 : 8) + r;
        float v = acc2[r];
        s1 += v * Wf[qcol * NE + eg];
        s2 += v * v;
      }
      atomicAdd(&w12[qcol], s1);
      atomicAdd(&w2sq[qcol], s2);
      __syncthreads();
    }
  }

  // ---------------- Phase 4: cosine -> exp-sum-log pooling ------------------
  if (tid < QP) {
    int q = tid;
    float n2 = 0.f;
    for (int e = 0; e < NE; ++e) { float w = Wf[q * NE + e]; n2 += w * w; }
    float w1 = sqrtf(n2);
    float w2 = sqrtf(w2sq[q]);
    float rs = w12[q] / fmaxf(w1 * w2, 1e-8f);
    red[q] = (q < NS && smask[q] > 0.5f) ? __expf(GAMMA2 * rs) : 0.f;
  }
  __syncthreads();
  if (tid == 0) {
    float sum = 0.f;
    for (int q = 0; q < QP; ++q) sum += red[q];
    simT[b_img * NB + i_sent] = GAMMA3 * logf(sum);  // transposed store
  }
}

// ---------------- sentence-level cosine matrix ------------------------------
extern "C" __global__ __launch_bounds__(64)
void damsm_global_sim_kernel(const float* __restrict__ cnn,
                             const float* __restrict__ rnn,
                             float* __restrict__ g) {
  int i = blockIdx.x, j = threadIdx.x;
  const float* ci = cnn + (size_t)i * NE;
  const float* rj = rnn + (size_t)j * NE;
  float dot = 0.f, ni = 0.f, nj = 0.f;
  for (int e = 0; e < NE; ++e) {
    float a = ci[e], b = rj[e];
    dot += a * b; ni += a * a; nj += b * b;
  }
  g[i * NB + j] = GAMMA3 * dot / fmaxf(sqrtf(ni) * sqrtf(nj), 1e-8f);
}

// ---------------- four cross-entropy scalars --------------------------------
__device__ __forceinline__ float lse_md_row(const float* M, int r) {
  float mx = -1e30f;
  for (int j = 0; j < NB; ++j) mx = fmaxf(mx, M[r * NB + j]);
  float s = 0.f;
  for (int j = 0; j < NB; ++j) s += __expf(M[r * NB + j] - mx);
  return mx + logf(s) - M[r * NB + r];
}
__device__ __forceinline__ float lse_md_col(const float* M, int c) {
  float mx = -1e30f;
  for (int j = 0; j < NB; ++j) mx = fmaxf(mx, M[j * NB + c]);
  float s = 0.f;
  for (int j = 0; j < NB; ++j) s += __expf(M[j * NB + c] - mx);
  return mx + logf(s) - M[c * NB + c];
}

extern "C" __global__ __launch_bounds__(64)
void damsm_ce_kernel(const float* __restrict__ simT,
                     const float* __restrict__ g,
                     float* __restrict__ out) {
  __shared__ float a0[NB], a1[NB], a2[NB], a3[NB];
  int i = threadIdx.x;
  a0[i] = lse_md_row(simT, i);  // w_loss0: rows of similarities (=simT)
  a1[i] = lse_md_col(simT, i);  // w_loss1: rows of similarities.T
  a2[i] = lse_md_row(g, i);     // s_loss0
  a3[i] = lse_md_col(g, i);     // s_loss1
  __syncthreads();
  if (i == 0) {
    float s0 = 0.f, s1 = 0.f, s2 = 0.f, s3 = 0.f;
    for (int k = 0; k < NB; ++k) { s0 += a0[k]; s1 += a1[k]; s2 += a2[k]; s3 += a3[k]; }
    out[0] = s0 / NB; out[1] = s1 / NB; out[2] = s2 / NB; out[3] = s3 / NB;
  }
}

// ---------------------------------------------------------------------------
extern "C" void kernel_launch(void* const* d_in, const int* in_sizes, int n_in,
                              void* d_out, int out_size, void* d_ws, size_t ws_size,
                              hipStream_t stream) {
  const float* cnn   = (const float*)d_in[0];
  const float* rnn   = (const float*)d_in[1];
  const float* img   = (const float*)d_in[2];
  const float* words = (const float*)d_in[3];
  const float* smask = (const float*)d_in[4];
  // d_in[5] = labels (arange(64)), implicit in the CE diagonal.

  // Workspace layout: simT[64*64] f32 | g[64*64] f32 | ctxT bf16 | ctxR bf16
  float*  simT = (float*)d_ws;
  float*  g    = simT + NB * NB;
  __bf16* ctxT = (__bf16*)((char*)d_ws + 32768);
  __bf16* ctxR = ctxT + (size_t)NB * SP * NE;
  float*  out  = (float*)d_out;

  (void)hipFuncSetAttribute((const void*)damsm_word_kernel,
                            hipFuncAttributeMaxDynamicSharedMemorySize,
                            SMEM_BYTES);

  damsm_pack_kernel<<<dim3(NB), dim3(256), 0, stream>>>(img, ctxT, ctxR);
  damsm_word_kernel<<<dim3(NB * NB), dim3(256), SMEM_BYTES, stream>>>(
      ctxT, ctxR, words, smask, simT);
  damsm_global_sim_kernel<<<dim3(NB), dim3(NB), 0, stream>>>(cnn, rnn, g);
  damsm_ce_kernel<<<dim3(1), dim3(NB), 0, stream>>>(simT, g, out);
}
